// Self_Attention_Muti_Head_74929999446279
// MI455X (gfx1250) — matrix operand order repaired and checked by reading, not compiled, not run on hardware
//
#include <hip/hip_runtime.h>
#include <hip/hip_bf16.h>

// MI455X / gfx1250 wave32 WMMA attention.
//  - All three GEMM stages run on v_wmma_f32_16x16x32_f16 (f16 in, f32 acc).
//  - 8192x8192 f16 score matrix (128MB) materialized in workspace: fits in
//    the 192MB global L2, so the two-pass softmax costs ~nothing in HBM.
//  - Kernel 2 stages K tiles into LDS with global_load_async_to_lds_b128
//    (ASYNCcnt), double-buffered, so the 4 waves of a block share one copy.
//  - Fragment layouts per CDNA5 ISA 7.12.2:
//      A (16x32 f16): lane m=lane&15, h=lane>>4 holds K = h*8+[0..7] and
//                     16+h*8+[0..7]  -> two contiguous 16B chunks row-major.
//      B (32x16 f16): lane n=lane&15, h=lane>>4 holds column n,
//                     K = h*8+[0..7] and 16+h*8+[0..7] (kernels 2-3 use K = h*16+[0..15] for A and B).
//      C (16x16 f32): VGPR r, lane l: row = r + 8*(l>>4), col = l&15.
//  - Workspace (~153MB): Qh[8192x512] f16, Kh[8192x512] f16 (== B^T for
//    Q@K^T), Vt[512x8192] f16 (== B^T for P@V), rowinv[8192] f32,
//    P[8192x8192] f16.

typedef __attribute__((ext_vector_type(16))) _Float16 v16h;
typedef __attribute__((ext_vector_type(8)))  _Float16 v8h;
typedef __attribute__((ext_vector_type(8)))  float    v8f;
typedef __attribute__((ext_vector_type(4)))  float    v4f;

#define N_TOK 8192
#define DIM   512
#define NHEAD 8
#define KBLK  32                             // keys staged per LDS buffer
#define SOFTMAX_SCALE 0.044194173824159216f  // 1/sqrt(512), applied AFTER softmax

__device__ __forceinline__ v8f wmma_f16(v16h a, v16h b, v8f c) {
  return __builtin_amdgcn_wmma_f32_16x16x32_f16(false, a, false, b, (short)0, c,
                                                false, false);
}

__device__ __forceinline__ v16h combine(v8h lo, v8h hi) {
  return __builtin_shufflevector(lo, hi, 0, 1, 2, 3, 4, 5, 6, 7,
                                 8, 9, 10, 11, 12, 13, 14, 15);
}

// A-fragment (16x32, M x K) from a row-major f16 matrix, leading dim `ld`.
__device__ __forceinline__ v16h load_a_frag(const _Float16* __restrict__ base, int ld) {
  const int lane = threadIdx.x & 31;
  const int m = lane & 15, h = lane >> 4;
  const _Float16* p = base + (size_t)m * ld + h * 16;
  v8h lo = *(const v8h*)(p);        // K = h*16    .. h*16+7
  v8h hi = *(const v8h*)(p + 8);    // K = h*16+8  .. h*16+15
  return combine(lo, hi);
}

// B-fragment (32x16, K x N) from a row-major *B-transposed* (N x K) matrix.
__device__ __forceinline__ v16h load_bt_frag(const _Float16* __restrict__ baseT, int ld) {
  const int lane = threadIdx.x & 31;
  const int n = lane & 15, h = lane >> 4;
  const _Float16* p = baseT + (size_t)n * ld + h * 16;
  v8h lo = *(const v8h*)(p);        // K = h*16    .. h*16+7
  v8h hi = *(const v8h*)(p + 8);    // K = h*16+8  .. h*16+15
  return combine(lo, hi);
}

// Async global->LDS b128 copy (ASYNCcnt). LDS VDST address = low 32 bits of
// the generic shared pointer (LDS aperture keeps the offset in addr[31:0]).
__device__ __forceinline__ void async_b128(const _Float16* g, _Float16* l) {
  const unsigned laddr = (unsigned)(unsigned long long)l;
  asm volatile("global_load_async_to_lds_b128 %0, %1, off"
               :: "v"(laddr), "v"(g) : "memory");
}

__device__ __forceinline__ void wait_async0() {
  asm volatile("s_wait_asynccnt 0" ::: "memory");
}

// ---------------------------------------------------------------------------
// Kernel 1: Q = xWq+bq, K = xWk+bk, V = xWv+bv  (f32 in -> f16 out)
// grid = (128 row-blocks, 16 col-pairs, 3 matrices), block = 128 (4 waves).
// ---------------------------------------------------------------------------
__global__ void qkv_proj_kernel(const float* __restrict__ x,
                                const float* __restrict__ Wq, const float* __restrict__ bq,
                                const float* __restrict__ Wk, const float* __restrict__ bk,
                                const float* __restrict__ Wv, const float* __restrict__ bv,
                                _Float16* __restrict__ Qh, _Float16* __restrict__ Kh,
                                _Float16* __restrict__ Vt) {
  const int wave = threadIdx.x >> 5;
  const int lane = threadIdx.x & 31;
  const int ln = lane & 15;
  const int h  = lane >> 4;
  const int row0 = blockIdx.x * 64 + wave * 16;
  const int col0 = blockIdx.y * 32;
  const int which = blockIdx.z;

  const float* __restrict__ W    = (which == 0) ? Wq : (which == 1) ? Wk : Wv;
  const float* __restrict__ bias = (which == 0) ? bq : (which == 1) ? bk : bv;
  const float b0s = bias[col0 + ln];
  const float b1s = bias[col0 + 16 + ln];

  v8f acc0 = {}; v8f acc1 = {};

  for (int d = 0; d < DIM; d += 32) {
    const float* xp = x + (size_t)(row0 + ln) * DIM + d + h * 8;
    v4f a0 = *(const v4f*)(xp);
    v4f a1 = *(const v4f*)(xp + 4);
    v4f a2 = *(const v4f*)(xp + 16);
    v4f a3 = *(const v4f*)(xp + 20);
    v16h a;
#pragma unroll
    for (int i = 0; i < 4; ++i) {
      a[i]      = (_Float16)a0[i];
      a[4 + i]  = (_Float16)a1[i];
      a[8 + i]  = (_Float16)a2[i];
      a[12 + i] = (_Float16)a3[i];
    }
    v16h bf0, bf1;
#pragma unroll
    for (int e = 0; e < 16; ++e) {
      const int k = d + ((e < 8) ? h * 8 + e : 16 + h * 8 + (e - 8));
      bf0[e] = (_Float16)W[(size_t)k * DIM + col0 + ln];
      bf1[e] = (_Float16)W[(size_t)k * DIM + col0 + 16 + ln];
    }
    acc0 = wmma_f16(a, bf0, acc0);
    acc1 = wmma_f16(a, bf1, acc1);
  }

  if (which < 2) {                    // Q, K: row-major f16
    _Float16* __restrict__ dst = (which == 0) ? Qh : Kh;
#pragma unroll
    for (int r = 0; r < 8; ++r) {
      const int row = row0 + 8 * h + r;
      dst[(size_t)row * DIM + col0 + ln]      = (_Float16)(acc0[r] + b0s);
      dst[(size_t)row * DIM + col0 + 16 + ln] = (_Float16)(acc1[r] + b1s);
    }
  } else {                            // V stored transposed: Vt[feature][token]
#pragma unroll
    for (int r = 0; r < 8; ++r) {
      const int row = row0 + 8 * h + r;
      Vt[(size_t)(col0 + ln) * N_TOK + row]      = (_Float16)(acc0[r] + b0s);
      Vt[(size_t)(col0 + 16 + ln) * N_TOK + row] = (_Float16)(acc1[r] + b1s);
    }
  }
}

// ---------------------------------------------------------------------------
// Kernel 2: S = Q@K^T, stable row softmax, P = exp(S - max),
// rowinv = scale / rowsum.  Block = 4 waves / 64 query rows; Q fragments stay
// resident in VGPRs (16 x v16h = 128 VGPRs per lane).  K tiles (32x512 f16 =
// 32KB) are staged into double-buffered LDS with async b128 loads so all 4
// waves share one global read; DMA of tile t+1 overlaps WMMA of tile t.
// ---------------------------------------------------------------------------
__global__ void softmax_scores_kernel(const _Float16* __restrict__ Qh,
                                      const _Float16* __restrict__ Kh,
                                      _Float16* __restrict__ P,
                                      float* __restrict__ rowinv) {
  __shared__ _Float16 kbuf[2][KBLK * DIM];   // 2 x 32KB of the WGP's 320KB

  const int tid  = threadIdx.x;
  const int wave = tid >> 5;
  const int lane = tid & 31;
  const int n = lane & 15, h = lane >> 4;
  const int row0 = blockIdx.x * 64 + wave * 16;

  v16h qf[16];
#pragma unroll
  for (int dc = 0; dc < 16; ++dc)
    qf[dc] = load_a_frag(Qh + (size_t)row0 * DIM + dc * 32, DIM);

  float mx[8];
#pragma unroll
  for (int r = 0; r < 8; ++r) mx[r] = -3.0e38f;

  // ---- Pass 1: raw scores -> P (f16), per-lane running max ----
  const int NB = N_TOK / KBLK;        // 256 staged key blocks
  // Stage block 0. A 32x512 f16 tile is contiguous in Kh: 2048 b128 chunks,
  // 16 per thread.
  {
    const _Float16* g = Kh;
#pragma unroll
    for (int i = 0; i < 16; ++i) {
      const int off = (tid + i * 128) * 8;
      async_b128(g + off, &kbuf[0][off]);
    }
  }
  wait_async0();
  __syncthreads();

  for (int jb = 0; jb < NB; ++jb) {
    const int b = jb & 1;
    if (jb + 1 < NB) {                // prefetch next tile into the other buffer
      const _Float16* g = Kh + (size_t)(jb + 1) * KBLK * DIM;
#pragma unroll
      for (int i = 0; i < 16; ++i) {
        const int off = (tid + i * 128) * 8;
        async_b128(g + off, &kbuf[b ^ 1][off]);
      }
    }
    const _Float16* kb = &kbuf[b][0];
#pragma unroll
    for (int kt = 0; kt < 2; ++kt) {  // two 16-key sub-tiles
      v8f s = {};
#pragma unroll
      for (int dc = 0; dc < 16; ++dc) {
        const _Float16* p = kb + (size_t)(kt * 16 + n) * DIM + dc * 32 + h * 16;
        v8h lo = *(const v8h*)(p);    // ds_load_b128
        v8h hi = *(const v8h*)(p + 8);
        s = wmma_f16(qf[dc], combine(lo, hi), s);
      }
      const int j0 = jb * KBLK + kt * 16;
#pragma unroll
      for (int r = 0; r < 8; ++r) {
        P[(size_t)(row0 + 8 * h + r) * N_TOK + j0 + n] = (_Float16)s[r];
        mx[r] = fmaxf(mx[r], s[r]);
      }
    }
    wait_async0();
    __syncthreads();
  }

  // Cross-lane row max (xor masks < 16 stay within each 16-lane half).
#pragma unroll
  for (int r = 0; r < 8; ++r) {
    float v = mx[r];
    for (int off = 1; off < 16; off <<= 1) v = fmaxf(v, __shfl_xor(v, off, 32));
    mx[r] = v;
  }

  // ---- Pass 2: P = exp(S - max) in-place (L2 hits), row sums ----
  float sum[8];
#pragma unroll
  for (int r = 0; r < 8; ++r) sum[r] = 0.0f;
  for (int j0 = 0; j0 < N_TOK; j0 += 16) {
#pragma unroll
    for (int r = 0; r < 8; ++r) {
      const size_t idx = (size_t)(row0 + 8 * h + r) * N_TOK + j0 + n;
      const float p = __expf((float)P[idx] - mx[r]);
      sum[r] += p;
      P[idx] = (_Float16)p;
    }
  }
#pragma unroll
  for (int r = 0; r < 8; ++r) {
    float v = sum[r];
    for (int off = 1; off < 16; off <<= 1) v += __shfl_xor(v, off, 32);
    if (n == 0) rowinv[row0 + 8 * h + r] = SOFTMAX_SCALE / v;  // scale AFTER softmax
  }
}

// ---------------------------------------------------------------------------
// Kernel 3: O = (P @ V) * rowinv, tiled x8 heads into the 8192x4096 output.
// grid = (128 row-blocks, 16 feature-pairs), block = 128 (4 waves).
// ---------------------------------------------------------------------------
__global__ void pv_out_kernel(const _Float16* __restrict__ P,
                              const _Float16* __restrict__ Vt,
                              const float* __restrict__ rowinv,
                              float* __restrict__ out) {
  const int wave = threadIdx.x >> 5;
  const int lane = threadIdx.x & 31;
  const int n = lane & 15, h = lane >> 4;
  const int row0 = blockIdx.x * 64 + wave * 16;
  const int f0 = blockIdx.y * 32;

  const _Float16* pA  = P  + (size_t)row0 * N_TOK;
  const _Float16* pB0 = Vt + (size_t)f0 * N_TOK;
  const _Float16* pB1 = Vt + (size_t)(f0 + 16) * N_TOK;

  v8f acc0 = {}; v8f acc1 = {};
  for (int j0 = 0; j0 < N_TOK; j0 += 32) {
    __builtin_prefetch((const void*)(pA + (size_t)n * N_TOK + j0 + 256), 0, 1);
    v16h a  = load_a_frag(pA + j0, N_TOK);
    v16h b0 = load_bt_frag(pB0 + j0, N_TOK);
    v16h b1 = load_bt_frag(pB1 + j0, N_TOK);
    acc0 = wmma_f16(a, b0, acc0);
    acc1 = wmma_f16(a, b1, acc1);
  }

#pragma unroll
  for (int r = 0; r < 8; ++r) {
    const int row = row0 + 8 * h + r;
    const float f = rowinv[row];
    const float o0 = acc0[r] * f;
    const float o1 = acc1[r] * f;
    float* orow = out + (size_t)row * (DIM * NHEAD);
#pragma unroll
    for (int head = 0; head < NHEAD; ++head) {   // all heads identical -> tile
      orow[head * DIM + f0 + n]      = o0;
      orow[head * DIM + f0 + 16 + n] = o1;
    }
  }
}

// ---------------------------------------------------------------------------
extern "C" void kernel_launch(void* const* d_in, const int* in_sizes, int n_in,
                              void* d_out, int out_size, void* d_ws, size_t ws_size,
                              hipStream_t stream) {
  const float* x  = (const float*)d_in[0];
  const float* Wq = (const float*)d_in[1];
  const float* bq = (const float*)d_in[2];
  const float* Wk = (const float*)d_in[3];
  const float* bk = (const float*)d_in[4];
  const float* Wv = (const float*)d_in[5];
  const float* bv = (const float*)d_in[6];
  // d_in[7] = nums_head (== 8 per setup_inputs; NHEAD is compile-time here)
  float* out = (float*)d_out;

  _Float16* Qh = (_Float16*)d_ws;
  _Float16* Kh = Qh + (size_t)N_TOK * DIM;
  _Float16* Vt = Kh + (size_t)N_TOK * DIM;
  float* rowinv = (float*)(Vt + (size_t)N_TOK * DIM);
  _Float16* P = (_Float16*)(rowinv + N_TOK);

  qkv_proj_kernel<<<dim3(N_TOK / 64, DIM / 32, 3), 128, 0, stream>>>(
      x, Wq, bq, Wk, bk, Wv, bv, Qh, Kh, Vt);
  softmax_scores_kernel<<<dim3(N_TOK / 64), 128, 0, stream>>>(Qh, Kh, P, rowinv);
  pv_out_kernel<<<dim3(N_TOK / 64, DIM / 32), 128, 0, stream>>>(P, Vt, rowinv, out);
}
